// LSTM_22290880266662
// MI455X (gfx1250) — matrix operand order repaired and checked by reading, not compiled, not run on hardware
//
#include <hip/hip_runtime.h>
#include <hip/hip_bf16.h>

// ---------------------------------------------------------------------------
// 2-layer LSTM (H=512, B=64, T=512, NF=64) + linear head for gfx1250 (CDNA5).
// fp16 WMMA (v_wmma_f32_16x16x32_f16) with fp32 accumulation.
// gates = [h | x_t] @ [W_hh | W_ih]^T + (b_ih + b_hh), fused per step.
// Persistent kernel, 32 WGs x 256 threads; weight slice staged once in LDS
// (via async global->LDS loads when available) in B-fragment layout and
// reused for all 512 steps; double-buffered h state; per-step device barrier
// with the x-projection GEMM overlapped into the barrier latency window.
// ---------------------------------------------------------------------------

typedef _Float16 h8   __attribute__((ext_vector_type(8)));
typedef _Float16 v16h __attribute__((ext_vector_type(16)));
typedef float    v8f  __attribute__((ext_vector_type(8)));

#define HSZ    512
#define BATCH  64
#define TSTEPS 512
#define G4H    2048
#define NWG    32      // HSZ / 16 hidden columns per workgroup

#if __has_builtin(__builtin_amdgcn_global_load_async_to_lds_b128)
#define HAVE_ASYNC_LDS 1
typedef int v4i_ __attribute__((ext_vector_type(4)));
typedef __attribute__((address_space(1))) v4i_ gv4i;   // global (AS1) int4
typedef __attribute__((address_space(3))) v4i_ lv4i;   // LDS (AS3) int4
#endif

// -------------------------------- aux kernels ------------------------------

__global__ void k_cvt_f32_to_f16(const float* __restrict__ src,
                                 _Float16* __restrict__ dst, int n) {
  int i = blockIdx.x * blockDim.x + threadIdx.x;
  if (i < n) dst[i] = (_Float16)src[i];
}

__global__ void k_bias_combine(const float* __restrict__ a,
                               const float* __restrict__ b,
                               float* __restrict__ out) {
  int i = blockIdx.x * blockDim.x + threadIdx.x;
  if (i < G4H) out[i] = a[i] + b[i];
}

// x: [B, T, 64] f32  ->  x0: [(t*64 + b)][64] f16
__global__ void k_pack_x(const float* __restrict__ x, _Float16* __restrict__ out) {
  int i = blockIdx.x * blockDim.x + threadIdx.x;   // over B*T*64
  if (i < BATCH * TSTEPS * 64) {
    int d  = i & 63;
    int bt = i >> 6;              // b*T + t
    int t  = bt & (TSTEPS - 1);
    int b  = bt >> 9;             // T = 512
    out[(size_t)((t << 6) + b) * 64 + d] = (_Float16)x[i];
  }
}

// zero both h-state buffers + reset sync counter
__global__ void k_zero_state(_Float16* __restrict__ hstate, unsigned* __restrict__ syncp) {
  int i = blockIdx.x * blockDim.x + threadIdx.x;
  if (i < 2 * BATCH * HSZ) hstate[i] = (_Float16)0.0f;
  if (i == 0) *syncp = 0u;
}

__device__ __forceinline__ float sigf(float x) {
  return 1.0f / (1.0f + __expf(-x));
}

// ----------------------------- recurrent kernel ----------------------------
// One workgroup per 16 hidden columns (hs = blockIdx.x*16).
// Waves 0..3: mtile = w, gates {i,f}, own c-state, do elementwise update.
// Waves 4..7: mtile = w-4, gates {g,o}, pass results via LDS exchange.
template <int KIN>
__global__ __launch_bounds__(256) void k_lstm_layer(
    const _Float16* __restrict__ Whh,    // [2048][512] f16
    const _Float16* __restrict__ Wih,    // [2048][KIN] f16
    const float*    __restrict__ bias,   // [2048] (b_ih + b_hh)
    const _Float16* __restrict__ xin,    // [T*64][KIN] f16, row = t*64+b
    _Float16*       __restrict__ hseq,   // [T*64][512] f16 out
    _Float16*       __restrict__ hstate, // [2][64][512] f16, pre-zeroed
    unsigned*       __restrict__ syncp)  // pre-zeroed
{
  constexpr int KTOT = HSZ + KIN;
  constexpr int KT   = KTOT / 32;        // total K-tiles of 32
  constexpr int KTH  = HSZ / 32;         // h-dependent K-tiles (16)
  constexpr int KTX  = KIN / 32;         // x-only K-tiles (2 or 16)

  __shared__ _Float16 bfrag[4 * KT * 32 * 16];   // B fragments: [gate][kt][lane][16]
  __shared__ float    xchg[2][4][16][16];        // g,o tiles: [g-2][mtile][row][col]

  const int tid   = threadIdx.x;
  const int lane  = tid & 31;
  const int wave  = tid >> 5;
  const int hs    = blockIdx.x * 16;
  const int mtile = wave & 3;
  const int ghalf = wave >> 2;     // 0 -> gates {0,1}=i,f ; 1 -> gates {2,3}=g,o
  const int g0    = ghalf * 2;
  const int g1    = g0 + 1;
  const int n     = lane & 15;
  const int hi16  = lane >> 4;
  const int jbase = hi16 * 8;      // A-fragment K sub-offset (0 or 8)

  // ---- stage weight slice into LDS in exact B-fragment order ----
  for (int s = tid; s < 4 * KT * 32; s += 256) {
    const int g   = s / (KT * 32);
    const int kt  = (s >> 5) % KT;
    const int l   = s & 31;
    const int nn  = l & 15;
    const int ko  = kt * 32 + ((l >> 4) * 8);    // K {ko..ko+7, ko+16..ko+23} per lane
    const int row = g * HSZ + hs + nn;           // gate column in 4H
    const _Float16* src = (ko < HSZ) ? (Whh + (size_t)row * HSZ + ko)
                                     : (Wih + (size_t)row * KIN + (ko - HSZ));
    _Float16* dst = &bfrag[(size_t)s * 16];
#ifdef HAVE_ASYNC_LDS
    __builtin_amdgcn_global_load_async_to_lds_b128(
        (gv4i*)src, (lv4i*)dst, 0, 0);
    __builtin_amdgcn_global_load_async_to_lds_b128(
        (gv4i*)(src + 16), (lv4i*)(dst + 8), 0, 0);
#else
    *(h8*)dst = *(const h8*)src;
    *(h8*)(dst + 8) = *(const h8*)(src + 16);
#endif
  }
#ifdef HAVE_ASYNC_LDS
#if __has_builtin(__builtin_amdgcn_s_wait_asynccnt)
  __builtin_amdgcn_s_wait_asynccnt(0);
#else
  asm volatile("s_wait_asynccnt 0" ::: "memory");
#endif
#endif
  __syncthreads();

  float cst[8];
  #pragma unroll
  for (int r = 0; r < 8; ++r) cst[r] = 0.0f;

  const float bias0 = bias[g0 * HSZ + hs + n];
  const float bias1 = bias[g1 * HSZ + hs + n];
  const int   Mb    = mtile * 16;
  const int   arow  = Mb + n;                    // this lane's A-matrix row

  _Float16* const hb0 = hstate;
  _Float16* const hb1 = hstate + (size_t)BATCH * HSZ;

  // ---- x-part prologue for t = 0 (bias + x_t @ W_ih^T) ----
  v8f xac0, xac1;
  {
    #pragma unroll
    for (int r = 0; r < 8; ++r) { xac0[r] = bias0; xac1[r] = bias1; }
    const _Float16* xrow = xin + (size_t)arow * KIN;
    #pragma unroll 4
    for (int kx = 0; kx < KTX; ++kx) {
      const int kt = KTH + kx;
      const _Float16* ap = xrow + kx * 32 + jbase;
      h8 lo = *(const h8*)ap;
      h8 hi = *(const h8*)(ap + 16);
      v16h a;
      #pragma unroll
      for (int q = 0; q < 8; ++q) { a[q] = lo[q]; a[8 + q] = hi[q]; }
      v16h b0v = *(const v16h*)&bfrag[(size_t)((g0 * KT + kt) * 32 + lane) * 16];
      v16h b1v = *(const v16h*)&bfrag[(size_t)((g1 * KT + kt) * 32 + lane) * 16];
      xac0 = __builtin_amdgcn_wmma_f32_16x16x32_f16(false, a, false, b0v,
                                                    (short)0, xac0, false, false);
      xac1 = __builtin_amdgcn_wmma_f32_16x16x32_f16(false, a, false, b1v,
                                                    (short)0, xac1, false, false);
    }
  }

  for (int t = 0; t < TSTEPS; ++t) {
    const _Float16* hprev = (t & 1) ? hb1 : hb0;
    _Float16*       hnext = (t & 1) ? hb0 : hb1;

    v8f acc0 = xac0, acc1 = xac1;

    // ---- h-dependent half of the GEMM (critical path) ----
    const _Float16* hrow = hprev + (size_t)arow * HSZ;
    #pragma unroll 4
    for (int kt = 0; kt < KTH; ++kt) {
      const _Float16* ap = hrow + kt * 32 + jbase;
      h8 lo = *(const h8*)ap;
      h8 hi = *(const h8*)(ap + 16);
      v16h a;
      #pragma unroll
      for (int q = 0; q < 8; ++q) { a[q] = lo[q]; a[8 + q] = hi[q]; }
      v16h b0v = *(const v16h*)&bfrag[(size_t)((g0 * KT + kt) * 32 + lane) * 16];
      v16h b1v = *(const v16h*)&bfrag[(size_t)((g1 * KT + kt) * 32 + lane) * 16];
      acc0 = __builtin_amdgcn_wmma_f32_16x16x32_f16(false, a, false, b0v,
                                                    (short)0, acc0, false, false);
      acc1 = __builtin_amdgcn_wmma_f32_16x16x32_f16(false, a, false, b1v,
                                                    (short)0, acc1, false, false);
    }

    if (ghalf == 1) {             // publish g,o tiles
      #pragma unroll
      for (int r = 0; r < 8; ++r) {
        const int rr = r + jbase;
        xchg[0][mtile][rr][n] = acc0[r];   // g gate
        xchg[1][mtile][rr][n] = acc1[r];   // o gate
      }
    }
    __syncthreads();

    if (ghalf == 0) {             // elementwise LSTM cell update
      #pragma unroll
      for (int r = 0; r < 8; ++r) {
        const int rr = r + jbase;
        const float iv = sigf(acc0[r]);
        const float fv = sigf(acc1[r]);
        const float gv = tanhf(xchg[0][mtile][rr][n]);
        const float ov = sigf(xchg[1][mtile][rr][n]);
        const float cn = fv * cst[r] + iv * gv;
        cst[r] = cn;
        const float hv = ov * tanhf(cn);
        const _Float16 hh = (_Float16)hv;
        const int M = Mb + rr;
        hnext[(size_t)M * HSZ + hs + n] = hh;
        hseq[(size_t)(t * BATCH + M) * HSZ + hs + n] = hh;
      }
    }
    __syncthreads();

    // signal: this WG's h(t) is globally visible
    if (tid == 0) {
      __threadfence();
      __hip_atomic_fetch_add(syncp, 1u, __ATOMIC_RELEASE, __HIP_MEMORY_SCOPE_AGENT);
    }

    // ---- overlap barrier latency: x-part GEMM for step t+1 ----
    if (t + 1 < TSTEPS) {
      #pragma unroll
      for (int r = 0; r < 8; ++r) { xac0[r] = bias0; xac1[r] = bias1; }
      const _Float16* xrow = xin + (size_t)((t + 1) * BATCH + arow) * KIN;
      if (t + 2 < TSTEPS)
        __builtin_prefetch(xin + (size_t)((t + 2) * BATCH + arow) * KIN, 0, 1);
      #pragma unroll 4
      for (int kx = 0; kx < KTX; ++kx) {
        const int kt = KTH + kx;
        const _Float16* ap = xrow + kx * 32 + jbase;
        h8 lo = *(const h8*)ap;
        h8 hi = *(const h8*)(ap + 16);
        v16h a;
        #pragma unroll
        for (int q = 0; q < 8; ++q) { a[q] = lo[q]; a[8 + q] = hi[q]; }
        v16h b0v = *(const v16h*)&bfrag[(size_t)((g0 * KT + kt) * 32 + lane) * 16];
        v16h b1v = *(const v16h*)&bfrag[(size_t)((g1 * KT + kt) * 32 + lane) * 16];
        xac0 = __builtin_amdgcn_wmma_f32_16x16x32_f16(false, a, false, b0v,
                                                      (short)0, xac0, false, false);
        xac1 = __builtin_amdgcn_wmma_f32_16x16x32_f16(false, a, false, b1v,
                                                      (short)0, xac1, false, false);
      }
    }

    // wait: all WGs' h(t) visible
    if (tid == 0) {
      const unsigned target = (unsigned)NWG * (unsigned)(t + 1);
      while (__hip_atomic_load(syncp, __ATOMIC_ACQUIRE, __HIP_MEMORY_SCOPE_AGENT) < target) {
        __builtin_amdgcn_s_sleep(1);
      }
    }
    __syncthreads();
    // make other WGs' h(t) stores visible to ALL waves of this WG
    __builtin_amdgcn_fence(__ATOMIC_ACQUIRE, "agent");
  }
}

// -------------------------------- head -------------------------------------
__global__ void k_head(const _Float16* __restrict__ h1,   // [T*64][512]
                       const float* __restrict__ Wfc,     // [512]
                       const float* __restrict__ bfc,     // [1]
                       float* __restrict__ out) {         // [B][T]
  int i = blockIdx.x * blockDim.x + threadIdx.x;          // over B*T
  if (i >= BATCH * TSTEPS) return;
  const int b = i / TSTEPS;
  const int t = i % TSTEPS;
  const _Float16* row = h1 + (size_t)(t * BATCH + b) * HSZ;
  float s = bfc[0];
  #pragma unroll 4
  for (int k = 0; k < HSZ; ++k) s += (float)row[k] * Wfc[k];
  out[i] = s; // out[b*T + t]
}

// ------------------------------ launcher ------------------------------------
extern "C" void kernel_launch(void* const* d_in, const int* in_sizes, int n_in,
                              void* d_out, int out_size, void* d_ws, size_t ws_size,
                              hipStream_t stream) {
  (void)in_sizes; (void)n_in; (void)out_size; (void)ws_size;

  const float* x     = (const float*)d_in[0];
  const float* W_ih0 = (const float*)d_in[1];
  const float* W_hh0 = (const float*)d_in[2];
  const float* b_ih0 = (const float*)d_in[3];
  const float* b_hh0 = (const float*)d_in[4];
  const float* W_ih1 = (const float*)d_in[5];
  const float* W_hh1 = (const float*)d_in[6];
  const float* b_ih1 = (const float*)d_in[7];
  const float* b_hh1 = (const float*)d_in[8];
  const float* W_fc  = (const float*)d_in[9];
  const float* b_fc  = (const float*)d_in[10];
  float* out = (float*)d_out;

  // workspace carve-up (256B aligned slabs)
  char* ws = (char*)d_ws;
  size_t off = 0;
  auto take = [&](size_t bytes) -> char* {
    char* p = ws + off;
    off += (bytes + 255) & ~(size_t)255;
    return p;
  };
  _Float16* x0     = (_Float16*)take((size_t)TSTEPS * BATCH * 64 * 2);
  _Float16* Wih0h  = (_Float16*)take((size_t)G4H * 64 * 2);
  _Float16* Whh0h  = (_Float16*)take((size_t)G4H * HSZ * 2);
  _Float16* Wih1h  = (_Float16*)take((size_t)G4H * HSZ * 2);
  _Float16* Whh1h  = (_Float16*)take((size_t)G4H * HSZ * 2);
  float*    bias0  = (float*)take((size_t)G4H * 4);
  float*    bias1  = (float*)take((size_t)G4H * 4);
  _Float16* h0seq  = (_Float16*)take((size_t)TSTEPS * BATCH * HSZ * 2);
  _Float16* h1seq  = (_Float16*)take((size_t)TSTEPS * BATCH * HSZ * 2);
  _Float16* hstate = (_Float16*)take((size_t)2 * BATCH * HSZ * 2);   // double buffer
  unsigned* syncp  = (unsigned*)take(256);

  // 1) convert weights to f16, combine biases, repack x
  k_cvt_f32_to_f16<<<(G4H * 64 + 255) / 256, 256, 0, stream>>>(W_ih0, Wih0h, G4H * 64);
  k_cvt_f32_to_f16<<<(G4H * HSZ + 255) / 256, 256, 0, stream>>>(W_hh0, Whh0h, G4H * HSZ);
  k_cvt_f32_to_f16<<<(G4H * HSZ + 255) / 256, 256, 0, stream>>>(W_ih1, Wih1h, G4H * HSZ);
  k_cvt_f32_to_f16<<<(G4H * HSZ + 255) / 256, 256, 0, stream>>>(W_hh1, Whh1h, G4H * HSZ);
  k_bias_combine<<<(G4H + 255) / 256, 256, 0, stream>>>(b_ih0, b_hh0, bias0);
  k_bias_combine<<<(G4H + 255) / 256, 256, 0, stream>>>(b_ih1, b_hh1, bias1);
  k_pack_x<<<(BATCH * TSTEPS * 64 + 255) / 256, 256, 0, stream>>>(x, x0);

  // 2) layer 0 (KIN = 64)
  k_zero_state<<<(2 * BATCH * HSZ + 255) / 256, 256, 0, stream>>>(hstate, syncp);
  k_lstm_layer<64><<<NWG, 256, 0, stream>>>(Whh0h, Wih0h, bias0, x0, h0seq, hstate, syncp);

  // 3) layer 1 (KIN = 512)
  k_zero_state<<<(2 * BATCH * HSZ + 255) / 256, 256, 0, stream>>>(hstate, syncp);
  k_lstm_layer<512><<<NWG, 256, 0, stream>>>(Whh1h, Wih1h, bias1, h0seq, h1seq, hstate, syncp);

  // 4) head
  k_head<<<(BATCH * TSTEPS + 255) / 256, 256, 0, stream>>>(h1seq, W_fc, b_fc, out);
}
